// ProteinMPNN_42107859370667
// MI455X (gfx1250) — matrix operand order repaired and checked by reading, not compiled, not run on hardware
//
#include <hip/hip_runtime.h>
#include <hip/hip_bf16.h>

// ---------------------------------------------------------------------------
// ProteinMPNN-style GNN on MI455X (gfx1250, wave32, WMMA bf16 path)
// All edge/node MLP matmuls run as v_wmma_f32_16x16x32_bf16 with f32 accum.
// Weights staged to LDS (padded pitch, bank-conflict free B-fragment reads),
// activations transposed between MLP layers through per-wave LDS scratch,
// segment_sum via global_atomic_add_f32, LayerNorm via wave32 shfl reductions.
// ---------------------------------------------------------------------------

#define HID     128
#define NNODES  10000
#define NEDGES  300000
#define NLAYERS 3
#define VOCAB   21
#define MSCALE  30.0f
#define LNEPS   1e-5f

typedef __attribute__((ext_vector_type(16))) __bf16 v16bf;
typedef __attribute__((ext_vector_type(8)))  float  v8f;

constexpr int LDP  = 136;  // padded LDS pitch for 128-wide rows (272B lane stride -> bank rotate)
constexpr int LDPF = 520;  // padded LDS pitch for 512-wide rows

__device__ __forceinline__ float gelu_exact(float x) {
  return 0.5f * x * (1.0f + erff(x * 0.70710678118654752440f));
}

__device__ __forceinline__ v8f wmma_bf16(v16bf a, v16bf b, v8f c) {
  // D = A(16x32) * B(32x16) + C, f32 accumulate
  return __builtin_amdgcn_wmma_f32_16x16x32_bf16(false, a, false, b, (short)0, c, false, false);
}

// A fragment: lane L holds row M=L&15; halves at K-chunks [c0,c0+8) and [c1,c1+8)
__device__ __forceinline__ v16bf frag_a(const __bf16* base, int c0, int c1) {
  union { uint4 q[2]; v16bf v; } u;
  u.q[0] = *(const uint4*)(base + c0);
  u.q[1] = *(const uint4*)(base + c1);
  return u.v;
}

// B fragment: lane L holds column N=L&15; K = 8h+s (s<8), 16+8h+(s-8) (s>=8), h=L>>4
__device__ __forceinline__ v16bf frag_b(const __bf16* w, int ld) {
  v16bf v;
  const int h = (threadIdx.x >> 4) & 1;
#pragma unroll
  for (int s = 0; s < 16; ++s) {
    const int k = (s < 8) ? (8 * h + s) : (16 + 8 * h + (s - 8));
    v[s] = w[(size_t)k * ld];
  }
  return v;
}

// cooperative copy of a [rows x 128] bf16 weight matrix into padded LDS
__device__ __forceinline__ void stage_weight(const __bf16* __restrict__ g,
                                             __bf16* __restrict__ s, int rows, int tid) {
  const int chunks = rows * 16;  // 16 uint4 (8 bf16 each) per row
  for (int i = tid; i < chunks; i += 256) {
    const int r = i >> 4, c = i & 15;
    ((uint4*)(s + r * LDP))[c] = ((const uint4*)(g + r * 128))[c];
  }
}

__device__ __forceinline__ float wave_sum(float v) {
#pragma unroll
  for (int m = 16; m >= 1; m >>= 1) v += __shfl_xor(v, m, 32);
  return v;
}

// ---------------------------------------------------------------------------
// Edge MLP3 (message or edge-update). KIN = 384 (enc) or 512 (dec message).
// EDGE_OUT=false: atomic-accumulate message into agg[row];  true: store he.
// Block = 256 threads = 8 waves; each wave owns a 16-edge tile.
// ---------------------------------------------------------------------------
template <int KIN, bool EDGE_OUT>
__global__ __launch_bounds__(256) void edge_mlp_kernel(
    const int* __restrict__ eidx, const __bf16* __restrict__ xb,
    const __bf16* __restrict__ eb, const __bf16* __restrict__ xlb,
    const float* __restrict__ mask,
    const __bf16* __restrict__ W1, const float* __restrict__ B1,
    const __bf16* __restrict__ W2, const float* __restrict__ B2,
    const __bf16* __restrict__ W3, const float* __restrict__ B3,
    float* __restrict__ agg, __bf16* __restrict__ he, int E, int nTiles) {
  extern __shared__ char smem[];
  __bf16* sW1  = (__bf16*)smem;           // [KIN x LDP]
  __bf16* sW2  = sW1 + (size_t)KIN * LDP; // [128 x LDP]
  __bf16* sW3  = sW2 + (size_t)128 * LDP; // [128 x LDP]
  __bf16* sAct = sW3 + (size_t)128 * LDP; // 8 waves * [16 x LDP]
  int*    sIdx = (int*)(sAct + (size_t)8 * 16 * LDP);

  const int tid  = threadIdx.x;
  const int wave = tid >> 5, lane = tid & 31;
  const int m = lane & 15, hi = lane >> 4;
  const int tile = blockIdx.x * 8 + wave;
  const bool active = (tile < nTiles);

  int row = 0, col = 0;
  if (active) {
    row = eidx[tile * 16 + m];
    col = eidx[E + tile * 16 + m];
    if (lane < 16) sIdx[wave * 16 + m] = row;  // row index per edge, for aggregation
  }

  stage_weight(W1, sW1, KIN, tid);
  stage_weight(W2, sW2, 128, tid);
  stage_weight(W3, sW3, 128, tid);
  __syncthreads();
  if (!active) return;

  const __bf16* seg0 = xb + (size_t)row * HID;
  const __bf16* seg1 = xb + (size_t)col * HID;
  const __bf16* seg2 = eb + (size_t)(tile * 16 + m) * HID;
  const __bf16* seg3 = xlb + (size_t)col * HID;
  float mmask = 0.f;
  if (KIN == 512) mmask = mask[tile * 16 + m];

  __bf16* act = sAct + (size_t)wave * 16 * LDP;
  const v8f vzero = {0.f, 0.f, 0.f, 0.f, 0.f, 0.f, 0.f, 0.f};
  v8f acc[8];

  // ---- GEMM1: [16 x KIN] x [KIN x 128] -------------------------------------
#pragma unroll
  for (int n = 0; n < 8; ++n) acc[n] = vzero;
  for (int ks = 0; ks < KIN / 32; ++ks) {
    const int kb = ks * 32;
    const int s = kb >> 7, off = kb & 127;
    const __bf16* src = (s == 0) ? seg0 : (s == 1) ? seg1 : (s == 2) ? seg2 : seg3;
    v16bf a = frag_a(src, off + hi * 8, off + 16 + hi * 8);
    if (KIN == 512 && s == 3) {
#pragma unroll
      for (int i = 0; i < 16; ++i) a[i] = (__bf16)((float)a[i] * mmask);
    }
#pragma unroll
    for (int n = 0; n < 8; ++n) {
      v16bf b = frag_b(sW1 + (size_t)kb * LDP + n * 16 + m, LDP);
      acc[n] = wmma_bf16(a, b, acc[n]);
    }
  }
  // bias + GELU, C-layout (M=r+8*hi, N=n*16+m) -> row-major LDS act tile
#pragma unroll
  for (int n = 0; n < 8; ++n) {
    const float bb = B1[n * 16 + m];
#pragma unroll
    for (int r = 0; r < 8; ++r)
      act[(r + 8 * hi) * LDP + n * 16 + m] = (__bf16)gelu_exact(acc[n][r] + bb);
  }

  // ---- GEMM2: [16 x 128] x [128 x 128] -------------------------------------
#pragma unroll
  for (int n = 0; n < 8; ++n) acc[n] = vzero;
  const __bf16* arow = act + (size_t)m * LDP;
  for (int ks = 0; ks < 4; ++ks) {
    const int kb = ks * 32;
    v16bf a = frag_a(arow, kb + hi * 8, kb + 16 + hi * 8);
#pragma unroll
    for (int n = 0; n < 8; ++n) {
      v16bf b = frag_b(sW2 + (size_t)kb * LDP + n * 16 + m, LDP);
      acc[n] = wmma_bf16(a, b, acc[n]);
    }
  }
#pragma unroll
  for (int n = 0; n < 8; ++n) {
    const float bb = B2[n * 16 + m];
#pragma unroll
    for (int r = 0; r < 8; ++r)
      act[(r + 8 * hi) * LDP + n * 16 + m] = (__bf16)gelu_exact(acc[n][r] + bb);
  }

  // ---- GEMM3: [16 x 128] x [128 x 128] -------------------------------------
#pragma unroll
  for (int n = 0; n < 8; ++n) acc[n] = vzero;
  for (int ks = 0; ks < 4; ++ks) {
    const int kb = ks * 32;
    v16bf a = frag_a(arow, kb + hi * 8, kb + 16 + hi * 8);
#pragma unroll
    for (int n = 0; n < 8; ++n) {
      v16bf b = frag_b(sW3 + (size_t)kb * LDP + n * 16 + m, LDP);
      acc[n] = wmma_bf16(a, b, acc[n]);
    }
  }

  if (!EDGE_OUT) {
    // segment_sum: atomic f32 accumulate into agg[row_of_edge]
#pragma unroll
    for (int n = 0; n < 8; ++n) {
      const float bb = B3[n * 16 + m];
#pragma unroll
      for (int r = 0; r < 8; ++r) {
        const int rr = sIdx[wave * 16 + r + 8 * hi];
        atomicAdd(agg + (size_t)rr * HID + n * 16 + m, acc[n][r] + bb);
      }
    }
  } else {
#pragma unroll
    for (int n = 0; n < 8; ++n) {
      const float bb = B3[n * 16 + m];
#pragma unroll
      for (int r = 0; r < 8; ++r)
        he[(size_t)(tile * 16 + r + 8 * hi) * HID + n * 16 + m] =
            (__bf16)(acc[n][r] + bb);
    }
  }
}

// ---------------------------------------------------------------------------
// Node FFN: out = GELU(x@Wd1 + bd1) @ Wd2 + bd2   (weights streamed from L2)
// ---------------------------------------------------------------------------
__global__ __launch_bounds__(256) void ffn_kernel(
    const __bf16* __restrict__ xb, const __bf16* __restrict__ Wd1,
    const float* __restrict__ Bd1, const __bf16* __restrict__ Wd2,
    const float* __restrict__ Bd2, float* __restrict__ out, int nTiles) {
  extern __shared__ char smem[];
  __bf16* sAct = (__bf16*)smem;  // 8 waves * [16 x LDPF]
  const int tid = threadIdx.x;
  const int wave = tid >> 5, lane = tid & 31;
  const int m = lane & 15, hi = lane >> 4;
  const int tile = blockIdx.x * 8 + wave;
  if (tile >= nTiles) return;

  const __bf16* arow = xb + (size_t)(tile * 16 + m) * HID;
  __bf16* h = sAct + (size_t)wave * 16 * LDPF;
  const v8f vzero = {0.f, 0.f, 0.f, 0.f, 0.f, 0.f, 0.f, 0.f};

  for (int chunk = 0; chunk < 4; ++chunk) {  // 512 outputs in 4 x 128
    v8f acc[8];
#pragma unroll
    for (int n = 0; n < 8; ++n) acc[n] = vzero;
    for (int ks = 0; ks < 4; ++ks) {
      const int kb = ks * 32;
      v16bf a = frag_a(arow, kb + hi * 8, kb + 16 + hi * 8);
#pragma unroll
      for (int n = 0; n < 8; ++n) {
        v16bf b = frag_b(Wd1 + (size_t)kb * 512 + chunk * 128 + n * 16 + m, 512);
        acc[n] = wmma_bf16(a, b, acc[n]);
      }
    }
#pragma unroll
    for (int n = 0; n < 8; ++n) {
      const float bb = Bd1[chunk * 128 + n * 16 + m];
#pragma unroll
      for (int r = 0; r < 8; ++r)
        h[(r + 8 * hi) * LDPF + chunk * 128 + n * 16 + m] =
            (__bf16)gelu_exact(acc[n][r] + bb);
    }
  }

  v8f acc[8];
#pragma unroll
  for (int n = 0; n < 8; ++n) acc[n] = vzero;
  const __bf16* hrow = h + (size_t)m * LDPF;
  for (int ks = 0; ks < 16; ++ks) {
    const int kb = ks * 32;
    v16bf a = frag_a(hrow, kb + hi * 8, kb + 16 + hi * 8);
#pragma unroll
    for (int n = 0; n < 8; ++n) {
      v16bf b = frag_b(Wd2 + (size_t)kb * HID + n * 16 + m, HID);
      acc[n] = wmma_bf16(a, b, acc[n]);
    }
  }
#pragma unroll
  for (int n = 0; n < 8; ++n) {
    const float bb = Bd2[n * 16 + m];
#pragma unroll
    for (int r = 0; r < 8; ++r)
      out[(size_t)(tile * 16 + r + 8 * hi) * HID + n * 16 + m] = acc[n][r] + bb;
  }
}

// ---------------------------------------------------------------------------
// x = LN(x + add*scale; g,b); also writes bf16 mirror. One wave per row.
// ---------------------------------------------------------------------------
__global__ __launch_bounds__(256) void residual_ln_kernel(
    float* __restrict__ x, const float* __restrict__ add, float scale,
    const float* __restrict__ g, const float* __restrict__ b,
    __bf16* __restrict__ xb, int rows) {
  const int wave = threadIdx.x >> 5, lane = threadIdx.x & 31;
  const int r = blockIdx.x * 8 + wave;
  if (r >= rows) return;
  const size_t base = (size_t)r * HID + lane * 4;
  float v[4], s = 0.f;
#pragma unroll
  for (int j = 0; j < 4; ++j) { v[j] = x[base + j] + add[base + j] * scale; s += v[j]; }
  s = wave_sum(s);
  const float mu = s * (1.0f / HID);
  float q = 0.f;
#pragma unroll
  for (int j = 0; j < 4; ++j) { const float d = v[j] - mu; q += d * d; }
  q = wave_sum(q);
  const float rs = rsqrtf(q * (1.0f / HID) + LNEPS);
#pragma unroll
  for (int j = 0; j < 4; ++j) {
    const int c = lane * 4 + j;
    const float y = g[c] * (v[j] - mu) * rs + b[c];
    x[base + j] = y;
    xb[base + j] = (__bf16)y;
  }
}

// e = LN(e + he; g,b), bf16 in/out. One wave per edge row.
__global__ __launch_bounds__(256) void edge_ln_kernel(
    __bf16* __restrict__ e, const __bf16* __restrict__ he,
    const float* __restrict__ g, const float* __restrict__ b, int rows) {
  const int wave = threadIdx.x >> 5, lane = threadIdx.x & 31;
  const int r = blockIdx.x * 8 + wave;
  if (r >= rows) return;
  const size_t base = (size_t)r * HID + lane * 4;
  float v[4], s = 0.f;
#pragma unroll
  for (int j = 0; j < 4; ++j) { v[j] = (float)e[base + j] + (float)he[base + j]; s += v[j]; }
  s = wave_sum(s);
  const float mu = s * (1.0f / HID);
  float q = 0.f;
#pragma unroll
  for (int j = 0; j < 4; ++j) { const float d = v[j] - mu; q += d * d; }
  q = wave_sum(q);
  const float rs = rsqrtf(q * (1.0f / HID) + LNEPS);
#pragma unroll
  for (int j = 0; j < 4; ++j) {
    const int c = lane * 4 + j;
    e[base + j] = (__bf16)(g[c] * (v[j] - mu) * rs + b[c]);
  }
}

__global__ void cvt_bf16_kernel(const float* __restrict__ s, __bf16* __restrict__ d, int n) {
  const int i = blockIdx.x * blockDim.x + threadIdx.x;
  if (i < n) d[i] = (__bf16)s[i];
}

__global__ void gather_label_kernel(const float* __restrict__ emb, const int* __restrict__ seq,
                                    __bf16* __restrict__ xlb, int total) {
  const int i = blockIdx.x * blockDim.x + threadIdx.x;
  if (i >= total) return;
  const int nidx = i >> 7, c = i & 127;
  xlb[i] = (__bf16)emb[seq[nidx] * HID + c];
}

__global__ void out_proj_kernel(const float* __restrict__ x, const float* __restrict__ W,
                                const float* __restrict__ b, float* __restrict__ out, int total) {
  const int i = blockIdx.x * blockDim.x + threadIdx.x;
  if (i >= total) return;
  const int nrow = i / VOCAB, v = i - nrow * VOCAB;
  float s = b[v];
  const float* xr = x + (size_t)nrow * HID;
#pragma unroll 8
  for (int c = 0; c < HID; ++c) s += xr[c] * W[c * VOCAB + v];
  out[i] = s;
}

// ---------------------------------------------------------------------------
extern "C" void kernel_launch(void* const* d_in, const int* in_sizes, int n_in,
                              void* d_out, int out_size, void* d_ws, size_t ws_size,
                              hipStream_t stream) {
  (void)in_sizes; (void)n_in; (void)out_size; (void)ws_size;
  // setup_inputs() insertion order; nested param dicts flattened in insertion order
  const float* in_x   = (const float*)d_in[0];
  const float* in_e   = (const float*)d_in[1];
  const float* in_lbl = (const float*)d_in[2];
  const float* eWv1 = (const float*)d_in[3];  const float* ebv1 = (const float*)d_in[4];
  const float* eWv2 = (const float*)d_in[5];  const float* ebv2 = (const float*)d_in[6];
  const float* eWv3 = (const float*)d_in[7];  const float* ebv3 = (const float*)d_in[8];
  const float* eg1  = (const float*)d_in[9];  const float* ebt1 = (const float*)d_in[10];
  const float* eg2  = (const float*)d_in[11]; const float* ebt2 = (const float*)d_in[12];
  const float* eWd1 = (const float*)d_in[13]; const float* ebd1 = (const float*)d_in[14];
  const float* eWd2 = (const float*)d_in[15]; const float* ebd2 = (const float*)d_in[16];
  const float* eWe1 = (const float*)d_in[17]; const float* ebe1 = (const float*)d_in[18];
  const float* eWe2 = (const float*)d_in[19]; const float* ebe2 = (const float*)d_in[20];
  const float* eWe3 = (const float*)d_in[21]; const float* ebe3 = (const float*)d_in[22];
  const float* eg3  = (const float*)d_in[23]; const float* ebt3 = (const float*)d_in[24];
  const float* dWv1 = (const float*)d_in[25]; const float* dbv1 = (const float*)d_in[26];
  const float* dWv2 = (const float*)d_in[27]; const float* dbv2 = (const float*)d_in[28];
  const float* dWv3 = (const float*)d_in[29]; const float* dbv3 = (const float*)d_in[30];
  const float* dg1  = (const float*)d_in[31]; const float* dbt1 = (const float*)d_in[32];
  const float* dg2  = (const float*)d_in[33]; const float* dbt2 = (const float*)d_in[34];
  const float* dWd1 = (const float*)d_in[35]; const float* dbd1 = (const float*)d_in[36];
  const float* dWd2 = (const float*)d_in[37]; const float* dbd2 = (const float*)d_in[38];
  const float* Wout = (const float*)d_in[39]; const float* bout = (const float*)d_in[40];
  const int*   eidx = (const int*)d_in[41];
  const int*   seq  = (const int*)d_in[42];
  const float* mask = (const float*)d_in[43];

  // ---- workspace carve ----
  char* wsb = (char*)d_ws;
  size_t off = 0;
  auto take = [&](size_t bytes) { size_t p = off; off += (bytes + 255) & ~(size_t)255; return p; };
  float*  xf  = (float*)(wsb + take((size_t)NNODES * HID * 4));
  float*  agg = (float*)(wsb + take((size_t)NNODES * HID * 4));
  __bf16* xb  = (__bf16*)(wsb + take((size_t)NNODES * HID * 2));
  __bf16* xlb = (__bf16*)(wsb + take((size_t)NNODES * HID * 2));
  __bf16* eb  = (__bf16*)(wsb + take((size_t)NEDGES * HID * 2));
  __bf16* he  = (__bf16*)(wsb + take((size_t)NEDGES * HID * 2));

  struct WSlot { const float* src; size_t n; __bf16* dst; };
  WSlot ws16[13];
  auto wtake = [&](const float* src, size_t n) {
    __bf16* p = (__bf16*)(wsb + take(n * 2));
    return WSlot{src, n, p};
  };
  ws16[0]  = wtake(eWv1, (size_t)NLAYERS * 384 * 128);
  ws16[1]  = wtake(eWv2, (size_t)NLAYERS * 128 * 128);
  ws16[2]  = wtake(eWv3, (size_t)NLAYERS * 128 * 128);
  ws16[3]  = wtake(eWe1, (size_t)NLAYERS * 384 * 128);
  ws16[4]  = wtake(eWe2, (size_t)NLAYERS * 128 * 128);
  ws16[5]  = wtake(eWe3, (size_t)NLAYERS * 128 * 128);
  ws16[6]  = wtake(eWd1, (size_t)NLAYERS * 128 * 512);
  ws16[7]  = wtake(eWd2, (size_t)NLAYERS * 512 * 128);
  ws16[8]  = wtake(dWv1, (size_t)NLAYERS * 512 * 128);
  ws16[9]  = wtake(dWv2, (size_t)NLAYERS * 128 * 128);
  ws16[10] = wtake(dWv3, (size_t)NLAYERS * 128 * 128);
  ws16[11] = wtake(dWd1, (size_t)NLAYERS * 128 * 512);
  ws16[12] = wtake(dWd2, (size_t)NLAYERS * 512 * 128);

  // ---- precompute: bf16 mirrors, f32 node state, label gather ----
  for (int i = 0; i < 13; ++i) {
    const int n = (int)ws16[i].n;
    cvt_bf16_kernel<<<(n + 255) / 256, 256, 0, stream>>>(ws16[i].src, ws16[i].dst, n);
  }
  {
    const int nx = NNODES * HID, ne = NEDGES * HID;
    cvt_bf16_kernel<<<(nx + 255) / 256, 256, 0, stream>>>(in_x, xb, nx);
    cvt_bf16_kernel<<<(ne + 255) / 256, 256, 0, stream>>>(in_e, eb, ne);
    gather_label_kernel<<<(nx + 255) / 256, 256, 0, stream>>>(in_lbl, seq, xlb, nx);
    hipMemcpyAsync(xf, in_x, (size_t)NNODES * HID * 4, hipMemcpyDeviceToDevice, stream);
  }

  const int eTiles = NEDGES / 16;          // 18750
  const int nTiles = NNODES / 16;          // 625
  const int eBlocks = (eTiles + 7) / 8;    // 2344
  const int nBlocks = (nTiles + 7) / 8;    // 79
  const int lnNodeBlocks = (NNODES + 7) / 8;
  const int lnEdgeBlocks = (NEDGES + 7) / 8;
  const size_t smemEnc = ((size_t)384 * LDP + 2 * 128 * LDP + 8 * 16 * LDP) * 2 + 8 * 16 * 4;
  const size_t smemDec = ((size_t)512 * LDP + 2 * 128 * LDP + 8 * 16 * LDP) * 2 + 8 * 16 * 4;
  const size_t smemFfn = (size_t)8 * 16 * LDPF * 2;

  // ---- encoder layers ----
  for (int l = 0; l < NLAYERS; ++l) {
    hipMemsetAsync(agg, 0, (size_t)NNODES * HID * 4, stream);
    edge_mlp_kernel<384, false><<<eBlocks, 256, smemEnc, stream>>>(
        eidx, xb, eb, xlb, mask,
        ws16[0].dst + (size_t)l * 384 * 128, ebv1 + l * 128,
        ws16[1].dst + (size_t)l * 128 * 128, ebv2 + l * 128,
        ws16[2].dst + (size_t)l * 128 * 128, ebv3 + l * 128,
        agg, he, NEDGES, eTiles);
    residual_ln_kernel<<<lnNodeBlocks, 256, 0, stream>>>(
        xf, agg, 1.0f / MSCALE, eg1 + l * 128, ebt1 + l * 128, xb, NNODES);
    ffn_kernel<<<nBlocks, 256, smemFfn, stream>>>(
        xb, ws16[6].dst + (size_t)l * 128 * 512, ebd1 + l * 512,
        ws16[7].dst + (size_t)l * 512 * 128, ebd2 + l * 128, agg, nTiles);
    residual_ln_kernel<<<lnNodeBlocks, 256, 0, stream>>>(
        xf, agg, 1.0f, eg2 + l * 128, ebt2 + l * 128, xb, NNODES);
    edge_mlp_kernel<384, true><<<eBlocks, 256, smemEnc, stream>>>(
        eidx, xb, eb, xlb, mask,
        ws16[3].dst + (size_t)l * 384 * 128, ebe1 + l * 128,
        ws16[4].dst + (size_t)l * 128 * 128, ebe2 + l * 128,
        ws16[5].dst + (size_t)l * 128 * 128, ebe3 + l * 128,
        agg, he, NEDGES, eTiles);
    edge_ln_kernel<<<lnEdgeBlocks, 256, 0, stream>>>(
        eb, he, eg3 + l * 128, ebt3 + l * 128, NEDGES);
  }

  // ---- decoder layers ----
  for (int l = 0; l < NLAYERS; ++l) {
    hipMemsetAsync(agg, 0, (size_t)NNODES * HID * 4, stream);
    edge_mlp_kernel<512, false><<<eBlocks, 256, smemDec, stream>>>(
        eidx, xb, eb, xlb, mask,
        ws16[8].dst + (size_t)l * 512 * 128, dbv1 + l * 128,
        ws16[9].dst + (size_t)l * 128 * 128, dbv2 + l * 128,
        ws16[10].dst + (size_t)l * 128 * 128, dbv3 + l * 128,
        agg, he, NEDGES, eTiles);
    residual_ln_kernel<<<lnNodeBlocks, 256, 0, stream>>>(
        xf, agg, 1.0f / MSCALE, dg1 + l * 128, dbt1 + l * 128, xb, NNODES);
    ffn_kernel<<<nBlocks, 256, smemFfn, stream>>>(
        xb, ws16[11].dst + (size_t)l * 128 * 512, dbd1 + l * 512,
        ws16[12].dst + (size_t)l * 512 * 128, dbd2 + l * 128, agg, nTiles);
    residual_ln_kernel<<<lnNodeBlocks, 256, 0, stream>>>(
        xf, agg, 1.0f, dg2 + l * 128, dbt2 + l * 128, xb, NNODES);
  }

  // ---- output projection ----
  const int total = NNODES * VOCAB;
  out_proj_kernel<<<(total + 255) / 256, 256, 0, stream>>>(
      xf, Wout, bout, (float*)d_out, total);
}
